// DoubleDCNv2plus_25288767439679
// MI455X (gfx1250) — matrix operand order repaired and checked by reading, not compiled, not run on hardware
//
#include <hip/hip_runtime.h>
#include <hip/hip_bf16.h>

typedef __attribute__((ext_vector_type(16))) __bf16 v16bf;
typedef __attribute__((ext_vector_type(8)))  float  v8f;

#define BSZ   8
#define CIN   64
#define COUT  128
#define HD    128
#define WD    128
#define HW    (HD * WD)
#define NPT   9
#define KTOT  576          // CIN * 9
#define KT    18           // KTOT / 32
#define CT    8            // COUT / 16
#define ASTRIDE 584        // 576 + 8 pad (keeps dword alignment, spreads banks)

static __device__ __forceinline__ unsigned short bf16_bits(float f) {
    unsigned int u = __float_as_uint(f);
    unsigned int lsb = (u >> 16) & 1u;
    u += 0x7fffu + lsb;                  // round-to-nearest-even
    return (unsigned short)(u >> 16);
}

// ---------------------------------------------------------------------------
// Zero the BN accumulator buffer (sum[128], sumsq[128]).
// ---------------------------------------------------------------------------
__global__ void init_sums_kernel(float* __restrict__ sums) {
    sums[threadIdx.x] = 0.0f;
}

// ---------------------------------------------------------------------------
// Pack W_conv (Cout,Cin,3,3) fp32 -> bf16 B-matrix fragments.
// Layout: [ct][kt][lane][e] with co = ct*16 + (lane&15), k = kt*32 + (e<8 ? 8*(lane>>4)+e : 16+8*(lane>>4)+e-8).
// The main kernel reads exactly this order, so A/B K-ordering is consistent.
// ---------------------------------------------------------------------------
__global__ void pack_w_kernel(const float* __restrict__ Wc,
                              unsigned short* __restrict__ bpack) {
    int idx = blockIdx.x * blockDim.x + threadIdx.x;
    if (idx >= CT * KT * 32 * 16) return;
    int e  = idx & 15;
    int L  = (idx >> 4) & 31;
    int kt = (idx >> 9) % KT;
    int ct = idx / (KT * 512);
    int co = ct * 16 + (L & 15);
    int k  = kt * 32 + ((e < 8) ? (8 * (L >> 4) + e) : (16 + 8 * (L >> 4) + (e - 8)));     // k = ci*9 + (kx*3+ky)
    bpack[idx] = bf16_bits(Wc[co * KTOT + k]);
}

// ---------------------------------------------------------------------------
// Offset + modulation convs (27 output channels, K=576), weights in LDS.
// One thread per (b,h,w). Sigmoid applied to the 9 modulation channels.
// ---------------------------------------------------------------------------
__global__ __launch_bounds__(256)
void offmod_kernel(const float* __restrict__ x,  const float* __restrict__ Wp,
                   const float* __restrict__ bp, const float* __restrict__ Wm,
                   const float* __restrict__ bm, float* __restrict__ offb,
                   float* __restrict__ mbuf) {
    __shared__ float lw[27 * KTOT];
    for (int i = threadIdx.x; i < 27 * KTOT; i += 256)
        lw[i] = (i < 18 * KTOT) ? Wp[i] : Wm[i - 18 * KTOT];
    __syncthreads();

    int idx = blockIdx.x * 256 + threadIdx.x;     // over B*H*W
    int b  = idx >> 14;
    int hw = idx & (HW - 1);
    int h  = hw >> 7, w = hw & 127;

    float acc[27];
#pragma unroll
    for (int oc = 0; oc < 27; ++oc) acc[oc] = 0.0f;

    const float* xb = x + (size_t)b * CIN * HW;
    for (int ci = 0; ci < CIN; ++ci) {
        const float* xc = xb + ci * HW;
#pragma unroll
        for (int kk = 0; kk < 9; ++kk) {
            int hh = h + kk / 3 - 1, ww = w + kk % 3 - 1;
            float xv = ((unsigned)hh < 128u && (unsigned)ww < 128u)
                           ? xc[hh * WD + ww] : 0.0f;
            int wi = ci * 9 + kk;
#pragma unroll
            for (int oc = 0; oc < 27; ++oc)
                acc[oc] = fmaf(xv, lw[oc * KTOT + wi], acc[oc]);
        }
    }
#pragma unroll
    for (int oc = 0; oc < 18; ++oc)
        offb[((b * 18 + oc) * HD + h) * WD + w] = acc[oc] + bp[oc];
#pragma unroll
    for (int oc = 0; oc < 9; ++oc) {
        float a = acc[18 + oc] + bm[oc];
        mbuf[((b * 9 + oc) * HD + h) * WD + w] = 1.0f / (1.0f + __expf(-a));
    }
}

// ---------------------------------------------------------------------------
// Fused deformable sampling + implicit GEMM (WMMA bf16) + BN stat accumulation.
// One block = one strip of 16 pixels (fixed b,h; w0..w0+15). 128 threads = 4 waves.
// Phase 1: stage modulated bilinear samples as a 16x576 bf16 A-tile in LDS.
// Phase 2: each wave owns cout tiles {wv, wv+4}; per k-step it loads the A
//          fragment ONCE and issues two independent WMMAs (A reuse, 2 accums).
// ---------------------------------------------------------------------------
__global__ __launch_bounds__(128)
void dcn_main_kernel(const float* __restrict__ x,
                     const float* __restrict__ offb,
                     const float* __restrict__ mbuf,
                     const unsigned short* __restrict__ bpack,
                     float* __restrict__ preBN,
                     float* __restrict__ sums) {
    __shared__ unsigned short lAs[16 * ASTRIDE];

    int bi  = blockIdx.x;          // 8192 strips
    int b   = bi >> 10;
    int rem = bi & 1023;
    int h   = rem >> 3;
    int w0  = (rem & 7) << 4;

    // ---------------- Phase 1: sample into LDS ----------------
    for (int pair = threadIdx.x; pair < 16 * NPT; pair += 128) {
        int p = pair / NPT, n = pair - p * NPT;
        int w = w0 + p;
        float offx = offb[((b * 18 + n) * HD + h) * WD + w];
        float offy = offb[((b * 18 + 9 + n) * HD + h) * WD + w];
        float mod  = mbuf[((b * 9 + n) * HD + h) * WD + w];
        float pnx = (float)(n / 3) - 1.0f;
        float pny = (float)(n % 3) - 1.0f;
        float px = offx + pnx + (float)(h + 1);   // padded-image coords
        float py = offy + pny + (float)(w + 1);
        float fx = floorf(px), fy = floorf(py);
        const float HI = 129.0f;                  // Hp-1
        float qltx = fminf(fmaxf(fx, 0.0f), HI);
        float qlty = fminf(fmaxf(fy, 0.0f), HI);
        float qrbx = fminf(fmaxf(fx + 1.0f, 0.0f), HI);
        float qrby = fminf(fmaxf(fy + 1.0f, 0.0f), HI);
        float pxc = fminf(fmaxf(px, 0.0f), HI);
        float pyc = fminf(fmaxf(py, 0.0f), HI);
        float wl = 1.0f + (qltx - pxc);
        float wr = 1.0f - (qrbx - pxc);
        float wt = 1.0f + (qlty - pyc);
        float wb = 1.0f - (qrby - pyc);
        float glt = wl * wt * mod, grb = wr * wb * mod;
        float glb = wl * wb * mod, grt = wr * wt * mod;
        // map padded coords -> unpadded x indices; zero-pad via weight kill
        int ax = (int)qltx - 1, ay = (int)qlty - 1;
        int cx = (int)qrbx - 1, cy = (int)qrby - 1;
        bool vax = (unsigned)ax < 128u, vay = (unsigned)ay < 128u;
        bool vcx = (unsigned)cx < 128u, vcy = (unsigned)cy < 128u;
        int iLT = (vax && vay) ? ax * WD + ay : 0; if (!(vax && vay)) glt = 0.0f;
        int iRB = (vcx && vcy) ? cx * WD + cy : 0; if (!(vcx && vcy)) grb = 0.0f;
        int iLB = (vax && vcy) ? ax * WD + cy : 0; if (!(vax && vcy)) glb = 0.0f;
        int iRT = (vcx && vay) ? cx * WD + ay : 0; if (!(vcx && vay)) grt = 0.0f;

        const float* xb = x + (size_t)b * CIN * HW;
        unsigned short* dst = &lAs[p * ASTRIDE + n];
        for (int ci = 0; ci < CIN; ++ci) {        // all 4 gathers hit 192MB L2
            const float* xc = xb + ci * HW;
            float v = glt * xc[iLT] + grb * xc[iRB] +
                      glb * xc[iLB] + grt * xc[iRT];
            dst[ci * 9] = bf16_bits(v);           // A[k = ci*9+n]
        }
    }
    __syncthreads();

    // ---------------- Phase 2: WMMA GEMM (2 cout tiles per wave) -----------
    int lane = threadIdx.x & 31;
    int wv   = threadIdx.x >> 5;
    int half = lane >> 4;
    int col  = lane & 15;
    int ct0  = wv;                 // cout tiles wv and wv+4
    int ct1  = wv + 4;

    union AU { v16bf v; unsigned int d[8]; };
    union BU { v16bf v; uint4 q[2]; };

    v8f acc0 = {};
    v8f acc1 = {};
#pragma unroll 2
    for (int kt = 0; kt < KT; ++kt) {
        AU a; BU b0, b1;
        int kb = kt * 32;
        // ISA 16-bit A layout: lane holds row M=lane&15; K pairs per VGPR,
        // lane-half selects K groups {0..7,16..23} vs {8..15,24..31}.
#pragma unroll
        for (int v = 0; v < 8; ++v) {
            int Kb = kb + ((v < 4) ? (2 * v + 8 * half)
                                   : (16 + 2 * (v - 4) + 8 * half));
            a.d[v] = *reinterpret_cast<const unsigned int*>(
                         lAs + col * ASTRIDE + Kb);
        }
        const uint4* g0 = reinterpret_cast<const uint4*>(
            bpack + (((size_t)(ct0 * KT + kt) * 32 + lane) << 4));
        const uint4* g1 = reinterpret_cast<const uint4*>(
            bpack + (((size_t)(ct1 * KT + kt) * 32 + lane) << 4));
        b0.q[0] = g0[0];  b0.q[1] = g0[1];
        b1.q[0] = g1[0];  b1.q[1] = g1[1];
        acc0 = __builtin_amdgcn_wmma_f32_16x16x32_bf16(
            false, a.v, false, b0.v, (short)0, acc0, false, false);
        acc1 = __builtin_amdgcn_wmma_f32_16x16x32_bf16(
            false, a.v, false, b1.v, (short)0, acc1, false, false);
    }

    // D layout: N = lane&15 (cout within tile), M = r + 8*half (pixel)
    {
        int co = ct0 * 16 + col;
        float s1 = 0.0f, s2 = 0.0f;
#pragma unroll
        for (int r = 0; r < 8; ++r) {
            float vv = acc0[r];
            int M = r + 8 * half;
            preBN[(((size_t)b * COUT + co) * HD + h) * WD + (w0 + M)] = vv;
            s1 += vv;
            s2 += vv * vv;
        }
        atomicAdd(&sums[co], s1);
        atomicAdd(&sums[COUT + co], s2);
    }
    {
        int co = ct1 * 16 + col;
        float s1 = 0.0f, s2 = 0.0f;
#pragma unroll
        for (int r = 0; r < 8; ++r) {
            float vv = acc1[r];
            int M = r + 8 * half;
            preBN[(((size_t)b * COUT + co) * HD + h) * WD + (w0 + M)] = vv;
            s1 += vv;
            s2 += vv * vv;
        }
        atomicAdd(&sums[co], s1);
        atomicAdd(&sums[COUT + co], s2);
    }
}

// ---------------------------------------------------------------------------
// Fold BN stats into per-channel scale/shift.
// ---------------------------------------------------------------------------
__global__ void bn_stats_kernel(const float* __restrict__ sums,
                                const float* __restrict__ gamma,
                                const float* __restrict__ beta,
                                float* __restrict__ stats) {
    int co = threadIdx.x;
    if (co < COUT) {
        const float inv = 1.0f / (float)(BSZ * HD * WD);
        float mean = sums[co] * inv;
        float var  = sums[COUT + co] * inv - mean * mean;
        float sA   = gamma[co] * rsqrtf(var + 1e-5f);
        stats[co]        = sA;
        stats[COUT + co] = beta[co] - mean * sA;
    }
}

// ---------------------------------------------------------------------------
// BN apply + ReLU, float4-vectorized (HBM bound: 134 MB moved).
// 4 consecutive elements always share a channel (stride 16384).
// ---------------------------------------------------------------------------
__global__ __launch_bounds__(256)
void bn_apply_kernel(const float4* __restrict__ preBN,
                     const float* __restrict__ stats,
                     float4* __restrict__ out) {
    int idx = blockIdx.x * 256 + threadIdx.x;           // over elements/4
    int co = (idx >> 12) & (COUT - 1);                  // (idx*4 >> 14) & 127
    float sA = stats[co];
    float sB = stats[COUT + co];
    float4 v = preBN[idx];
    float4 o;
    o.x = fmaxf(v.x * sA + sB, 0.0f);
    o.y = fmaxf(v.y * sA + sB, 0.0f);
    o.z = fmaxf(v.z * sA + sB, 0.0f);
    o.w = fmaxf(v.w * sA + sB, 0.0f);
    out[idx] = o;
}

// ---------------------------------------------------------------------------
extern "C" void kernel_launch(void* const* d_in, const int* in_sizes, int n_in,
                              void* d_out, int out_size, void* d_ws, size_t ws_size,
                              hipStream_t stream) {
    const float* x     = (const float*)d_in[0];
    const float* Wp    = (const float*)d_in[1];
    const float* bp    = (const float*)d_in[2];
    const float* Wm    = (const float*)d_in[3];
    const float* bm    = (const float*)d_in[4];
    const float* Wc    = (const float*)d_in[5];
    const float* gamma = (const float*)d_in[6];
    const float* beta  = (const float*)d_in[7];
    float* out = (float*)d_out;

    char* ws = (char*)d_ws;
    const size_t OFF_SZ = (size_t)BSZ * 18 * HW * sizeof(float);  // 9.4 MB
    const size_t M_SZ   = (size_t)BSZ * 9 * HW * sizeof(float);   // 4.7 MB
    const size_t BP_SZ  = (size_t)CT * KT * 32 * 16 * sizeof(unsigned short);
    const size_t PRE_SZ = (size_t)BSZ * COUT * HW * sizeof(float);// 67 MB

    float*          offb  = (float*)ws;
    float*          mbuf  = (float*)(ws + OFF_SZ);
    unsigned short* bpack = (unsigned short*)(ws + OFF_SZ + M_SZ);
    float*          preBN = (float*)(ws + OFF_SZ + M_SZ + BP_SZ);
    float*          sums  = (float*)(ws + OFF_SZ + M_SZ + BP_SZ + PRE_SZ);
    float*          stats = sums + 2 * COUT;

    init_sums_kernel<<<1, 256, 0, stream>>>(sums);
    pack_w_kernel<<<(CT * KT * 512 + 255) / 256, 256, 0, stream>>>(Wc, bpack);
    offmod_kernel<<<(BSZ * HW) / 256, 256, 0, stream>>>(x, Wp, bp, Wm, bm, offb, mbuf);
    dcn_main_kernel<<<BSZ * HD * (WD / 16), 128, 0, stream>>>(x, offb, mbuf, bpack, preBN, sums);
    bn_stats_kernel<<<1, 128, 0, stream>>>(sums, gamma, beta, stats);
    bn_apply_kernel<<<(BSZ * COUT * HW) / 1024, 256, 0, stream>>>(
        (const float4*)preBN, stats, (float4*)out);
}
